// MyLSTM_3728031613796
// MI455X (gfx1250) — matrix operand order repaired and checked by reading, not compiled, not run on hardware
//
#include <hip/hip_runtime.h>
#include <math.h>

// ---------------------------------------------------------------------------
// LSTM forward (B=64, T=512, D=H=512), return_sequences.
// One persistent kernel; per step fused GEMM
//   z_t = [x_t | h_{t-1}] (64x1024 bf16) @ WU^T (1024x2048 bf16) + b
// via v_wmma_f32_16x16x32_bf16. Double-buffered LDS tiles with async
// global->LDS copies (ASYNCcnt) where data is already bf16; grid-wide
// barrier once per timestep. Weights (4MB bf16) stay L2-resident.
// ---------------------------------------------------------------------------

typedef __attribute__((ext_vector_type(16))) __bf16 v16bf;
typedef __attribute__((ext_vector_type(8)))  __bf16 v8bf;
typedef __attribute__((ext_vector_type(8)))  float  v8f;
typedef __attribute__((ext_vector_type(4)))  int    v4i;

static constexpr int Bq   = 64;     // batch
static constexpr int Tt   = 512;    // time steps
static constexpr int Dd   = 512;    // input dim
static constexpr int Hh   = 512;    // hidden dim
static constexpr int KTOT = 1024;   // D + H fused K
static constexpr int NGATE_COLS = 4 * Hh;  // 2048
static constexpr int NWG  = 32;     // workgroups; each owns 16 h-cols
static constexpr int NTHR = 512;    // 16 wave32 per WG
static constexpr int ASTR = 136;    // LDS A row stride (bf16), 272B rows
static constexpr int BSTR = 136;    // LDS B row stride (bf16)

#if __has_builtin(__builtin_amdgcn_global_load_async_to_lds_b128)
#define ASYNC_LDS 1
#else
#define ASYNC_LDS 0
#endif

typedef __attribute__((address_space(1))) v4i gv4i;
typedef __attribute__((address_space(3))) v4i lv4i;

__device__ __forceinline__ void async_copy16(void* lds_dst, const void* gsrc) {
#if ASYNC_LDS
  __builtin_amdgcn_global_load_async_to_lds_b128(
      (gv4i*)gsrc, (lv4i*)lds_dst, 0 /*offset*/, 0 /*cpol*/);
#else
  *(v8bf*)lds_dst = *(const v8bf*)gsrc;
#endif
}

__device__ __forceinline__ void async_wait0() {
#if ASYNC_LDS
#if __has_builtin(__builtin_amdgcn_s_wait_asynccnt)
  __builtin_amdgcn_s_wait_asynccnt(0);
#else
  asm volatile("s_wait_asynccnt 0" ::: "memory");
#endif
#endif
}

__device__ __forceinline__ float sigmoidf_(float x) {
  return 1.0f / (1.0f + __expf(-x));
}

// Pack [W;U] transposed to bf16: WUT[n][k], n in [0,2048), k in [0,1024).
__global__ void lstm_pack_wu(const float* __restrict__ Wk,
                             const float* __restrict__ Uk,
                             __bf16* __restrict__ WUT) {
  int idx = blockIdx.x * blockDim.x + threadIdx.x;
  if (idx >= NGATE_COLS * KTOT) return;
  int k = idx & (KTOT - 1);
  int n = idx >> 10;
  float v = (k < Dd) ? Wk[(size_t)k * NGATE_COLS + n]
                     : Uk[(size_t)(k - Dd) * NGATE_COLS + n];
  WUT[(size_t)n * KTOT + k] = (__bf16)v;
}

__global__ void __launch_bounds__(NTHR)
lstm_persistent(const float* __restrict__ x,      // [B, T, D] fp32
                const float* __restrict__ bias,   // [4H] fp32
                const __bf16* __restrict__ WUT,   // [2048][1024] bf16
                __bf16* __restrict__ hbf,         // [2][B][H] bf16 ping-pong
                float*  __restrict__ cbuf,        // [B][H] fp32 (activated c)
                unsigned* __restrict__ barCount,
                unsigned* __restrict__ barGen,
                float* __restrict__ out) {        // [B, T, H] fp32
  __shared__ __bf16 As[2][Bq][ASTR];     // double-buffered A chunk
  __shared__ __bf16 Bs[2][64][BSTR];     // double-buffered B chunk
  __shared__ float  Zs[4][Bq][16];       // pre-activation z tiles
  __shared__ float  bias_s[4][16];

  const int tid  = threadIdx.x;
  const int blk  = blockIdx.x;           // owns h-cols [blk*16, blk*16+16)
  const int lane = tid & 31;
  const int wv   = tid >> 5;             // 0..15
  const int m_tile = wv & 3;             // which 16 batch rows
  const int gate   = wv >> 2;            // i/f/c/o

  if (tid < 64)
    bias_s[tid >> 4][tid & 15] = bias[(tid >> 4) * Hh + blk * 16 + (tid & 15)];

  // cooperative-load mapping: thread -> (row 0..63, 16-elem segment 0..7)
  const int ldrow = tid >> 3;
  const int ldj   = (tid & 7) * 16;
  const int nl = ldrow;                  // B tile LDS row
  const size_t nglob = (size_t)((nl >> 4) * Hh + blk * 16 + (nl & 15));

  // fragment addressing per ISA 7.12.2 (16-bit A 16x32 / B 32x16 layouts)
  const int arow = m_tile * 16 + (lane & 15);
  const int klo  = (lane >> 4) * 8;
  const int bn   = gate * 16 + (lane & 15);
  const int bko  = (lane >> 4) * 8;

  for (int t = 0; t < Tt; ++t) {
    v8f acc = {};
    const __bf16* hread = hbf + (size_t)(t & 1) * (Bq * Hh);

    // stage K-chunk `kc` (128 deep) into LDS buffer `bufi`
    auto stage = [&](int kc, int bufi) {
      // ---- A: x_t (fp32 -> bf16 manual) or h_{t-1} (bf16, async) ----
      if (kc < 4) {
        const float4* src =
            (const float4*)(x + ((size_t)ldrow * Tt + t) * Dd + kc * 128 + ldj);
        float4 f0 = src[0], f1 = src[1], f2 = src[2], f3 = src[3];
        v8bf o0, o1;
        o0[0]=(__bf16)f0.x; o0[1]=(__bf16)f0.y; o0[2]=(__bf16)f0.z; o0[3]=(__bf16)f0.w;
        o0[4]=(__bf16)f1.x; o0[5]=(__bf16)f1.y; o0[6]=(__bf16)f1.z; o0[7]=(__bf16)f1.w;
        o1[0]=(__bf16)f2.x; o1[1]=(__bf16)f2.y; o1[2]=(__bf16)f2.z; o1[3]=(__bf16)f2.w;
        o1[4]=(__bf16)f3.x; o1[5]=(__bf16)f3.y; o1[6]=(__bf16)f3.z; o1[7]=(__bf16)f3.w;
        *(v8bf*)&As[bufi][ldrow][ldj]     = o0;
        *(v8bf*)&As[bufi][ldrow][ldj + 8] = o1;
      } else if (t == 0) {               // h_{-1} == 0
        v8bf z = {};
        *(v8bf*)&As[bufi][ldrow][ldj]     = z;
        *(v8bf*)&As[bufi][ldrow][ldj + 8] = z;
      } else {
        const __bf16* src = hread + (size_t)ldrow * Hh + (kc - 4) * 128 + ldj;
        async_copy16(&As[bufi][ldrow][ldj],     src);
        async_copy16(&As[bufi][ldrow][ldj + 8], src + 8);
      }
      // ---- B: WU^T chunk, already bf16, K-major coalesced, async ----
      const __bf16* wsrc = WUT + nglob * KTOT + kc * 128 + ldj;
      async_copy16(&Bs[bufi][nl][ldj],     wsrc);
      async_copy16(&Bs[bufi][nl][ldj + 8], wsrc + 8);
      if (kc < 7)
        __builtin_prefetch(WUT + nglob * KTOT + (kc + 1) * 128 + ldj, 0, 3);
    };

    // prologue: fill buffer 0 with chunk 0
    stage(0, 0);
    async_wait0();
    __syncthreads();

    for (int kc = 0; kc < 8; ++kc) {
      const int cur = kc & 1;
      if (kc < 7) stage(kc + 1, cur ^ 1);   // overlap copy with compute

      // ---- 4 x v_wmma_f32_16x16x32_bf16 on the current 128-deep chunk ----
#pragma unroll
      for (int ks = 0; ks < 128; ks += 32) {
        const __bf16* ap = &As[cur][arow][ks + klo];
        v8bf alo = *(const v8bf*)ap;
        v8bf ahi = *(const v8bf*)(ap + 16);
        const __bf16* bp = &Bs[cur][bn][ks + bko];
        v8bf blo = *(const v8bf*)bp;
        v8bf bhi = *(const v8bf*)(bp + 16);
        v16bf af, bfr;
#pragma unroll
        for (int i = 0; i < 8; ++i) {
          af[i] = alo[i]; af[8 + i] = ahi[i];
          bfr[i] = blo[i]; bfr[8 + i] = bhi[i];
        }
        acc = __builtin_amdgcn_wmma_f32_16x16x32_bf16(
            false, af, false, bfr, (short)0, acc, false, false);
      }

      if (kc < 7) async_wait0();   // next chunk's async copies complete
      __syncthreads();             // flip buffers
    }

    // ---- spill z tiles to LDS (C/D layout: vgpr r, lane<16 -> M=r else r+8)
    {
      const int roff = m_tile * 16 + ((lane >> 4) << 3);
      const int col  = lane & 15;
#pragma unroll
      for (int r = 0; r < 8; ++r) Zs[gate][roff + r][col] = acc[r];
    }
    __syncthreads();

    // ---- gate nonlinearity + state update for this WG's 16 h-cols ----
    __bf16* hwrite = hbf + (size_t)((t + 1) & 1) * (Bq * Hh);
#pragma unroll
    for (int e = tid; e < Bq * 16; e += NTHR) {
      int row  = e >> 4;
      int col  = e & 15;
      int colg = blk * 16 + col;
      float zi = Zs[0][row][col] + bias_s[0][col];
      float zf = Zs[1][row][col] + bias_s[1][col];
      float zc = Zs[2][row][col] + bias_s[2][col];
      float zo = Zs[3][row][col] + bias_s[3][col];
      float cp = (t == 0) ? 0.0f : cbuf[row * Hh + colg];
      // reference carries the ACTIVATED cell state
      float cn = tanhf(sigmoidf_(zf) * cp + sigmoidf_(zi) * tanhf(zc));
      float hn = sigmoidf_(zo) * cn;
      cbuf[row * Hh + colg] = cn;
      out[((size_t)row * Tt + t) * Hh + colg] = hn;
      hwrite[row * Hh + colg] = (__bf16)hn;
    }

    // ---- grid-wide sense-reversing barrier (one per timestep) ----
    __threadfence();
    __syncthreads();
    if (tid == 0) {
      unsigned g   = *((volatile unsigned*)barGen);
      unsigned old = atomicAdd(barCount, 1u);
      if (old == NWG - 1) {
        *((volatile unsigned*)barCount) = 0u;
        __threadfence();
        atomicAdd(barGen, 1u);
      } else {
        while (*((volatile unsigned*)barGen) == g) __builtin_amdgcn_s_sleep(2);
      }
      __threadfence();
    }
    __syncthreads();
  }
}

extern "C" void kernel_launch(void* const* d_in, const int* in_sizes, int n_in,
                              void* d_out, int out_size, void* d_ws, size_t ws_size,
                              hipStream_t stream) {
  (void)in_sizes; (void)n_in; (void)out_size; (void)ws_size;
  const float* x    = (const float*)d_in[0];   // [64,512,512]
  const float* Wk   = (const float*)d_in[1];   // [512,2048]
  const float* Uk   = (const float*)d_in[2];   // [512,2048]
  const float* bias = (const float*)d_in[3];   // [2048]
  float* out = (float*)d_out;                  // [64,512,512]

  // workspace layout (needs ~4.5 MB)
  char* ws = (char*)d_ws;
  __bf16*   WUT  = (__bf16*)ws;                               // 4 MiB
  __bf16*   hbf  = (__bf16*)(ws + (4u << 20));                // 128 KiB (x2 buf)
  float*    cbuf = (float*)(ws + (4u << 20) + (128u << 10));  // 128 KiB
  unsigned* bar  = (unsigned*)(ws + (4u << 20) + (256u << 10));

  lstm_pack_wu<<<(NGATE_COLS * KTOT) / 256, 256, 0, stream>>>(Wk, Uk, WUT);
  (void)hipMemsetAsync(bar, 0, 128, stream);  // barrier words, capture-legal
  lstm_persistent<<<NWG, NTHR, 0, stream>>>(x, bias, WUT, hbf, cbuf,
                                            bar, bar + 1, out);
}
